// MultiHeadAttention_76020921139797
// MI455X (gfx1250) — matrix operand order repaired and checked by reading, not compiled, not run on hardware
//
#include <hip/hip_runtime.h>
#include <hip/hip_bf16.h>
#include <stdint.h>

// ---------------------------------------------------------------------------
// MHA forward for MI455X (gfx1250, wave32, WMMA bf16 16x16x32, f32 accum)
// B=2 S=2048 D=1024 H=16 DK=64
// All GEMMs run on v_wmma_f32_16x16x32_bf16; inputs/weights pre-converted to
// bf16 once so GEMM inner loops are pure b128 loads + WMMA.
// ---------------------------------------------------------------------------

typedef __attribute__((ext_vector_type(16))) __bf16 v16bf;
typedef __attribute__((ext_vector_type(8)))  __bf16 v8bf;
typedef __attribute__((ext_vector_type(4)))  __bf16 v4bf;
typedef __attribute__((ext_vector_type(8)))  float  v8f;
typedef __attribute__((ext_vector_type(4)))  float  f32x4;

static constexpr int Bc  = 2;
static constexpr int Sc  = 2048;
static constexpr int Dc  = 1024;
static constexpr int Hc  = 16;
static constexpr int DKc = 64;

__device__ __forceinline__ __bf16 f2bf(float f) {
  unsigned int u = __builtin_bit_cast(unsigned int, f);
  unsigned int r = u + 0x7FFFu + ((u >> 16) & 1u);   // round-to-nearest-even
  unsigned short s = (unsigned short)(r >> 16);
  return __builtin_bit_cast(__bf16, s);
}

// 16 bf16 from two contiguous 8-element chunks (each 16B aligned)
__device__ __forceinline__ v16bf ldbf(const __bf16* p1, const __bf16* p2) {
  v8bf lo = *(const v8bf*)p1;
  v8bf hi = *(const v8bf*)p2;
  return __builtin_shufflevector(lo, hi, 0,1,2,3,4,5,6,7,8,9,10,11,12,13,14,15);
}

// 16 bf16 converted from two contiguous 8-float chunks (LDS probs in attn)
__device__ __forceinline__ v16bf cvtbf(const float* p1, const float* p2) {
  v16bf a;
#pragma unroll
  for (int i = 0; i < 8; ++i) { a[i] = f2bf(p1[i]); a[8 + i] = f2bf(p2[i]); }
  return a;
}

__device__ __forceinline__ v8f wmma_bf16(v16bf a, v16bf b, v8f c) {
  return __builtin_amdgcn_wmma_f32_16x16x32_bf16(false, a, false, b, (short)0, c,
                                                 false, false);
}

// ---------------------------------------------------------------------------
// fp32 -> bf16 bulk conversion (one float4 -> one v4bf per thread)
// ---------------------------------------------------------------------------
__global__ __launch_bounds__(256) void cvt_bf16_kernel(
    const float* __restrict__ src, __bf16* __restrict__ dst) {
  const size_t i = (size_t)blockIdx.x * 256 + threadIdx.x;
  f32x4 v = ((const f32x4*)src)[i];
  v4bf o;
  o[0] = f2bf(v.x); o[1] = f2bf(v.y); o[2] = f2bf(v.z); o[3] = f2bf(v.w);
  ((v4bf*)dst)[i] = o;
}

// ---------------------------------------------------------------------------
// GEMM: P[m][n] = sum_k X[m][k] * W[n][k] + bias[n]
//   M = B*S = 4096, N = D = 1024, K = D = 1024 (X, W both bf16 row-major)
// MODE 0: out bf16, head-major   (b,h,s,dk)  -> Q / K
// MODE 1: out bf16, head-transp  (b,h,dk,s)  -> V (attn@V B-operand layout)
// MODE 2: out f32,  row-major    (m,n)       -> final output projection
// Wave computes a 16x64 tile (4 accumulators). 8 waves / block.
// ---------------------------------------------------------------------------
template <int MODE>
__global__ __launch_bounds__(256) void mha_gemm_kernel(
    const __bf16* __restrict__ X, const __bf16* __restrict__ W,
    const float* __restrict__ bias, void* __restrict__ outp) {
  constexpr int K = Dc, N = Dc;
  const int lane = threadIdx.x & 31;
  const int w    = threadIdx.x >> 5;
  const int gw   = blockIdx.x * 8 + w;         // 0..4095
  const int mt   = gw >> 4;                    // 0..255  (16-row tile)
  const int n0   = (gw & 15) * 64;             // column base
  const int l16  = lane & 15;
  const int hi   = lane >> 4;                  // 0 or 1

  const __bf16* arow = X + (size_t)(mt * 16 + l16) * K;
  const __bf16* wrow0 = W + (size_t)(n0 + l16) * K;

  v8f acc[4];
#pragma unroll
  for (int t = 0; t < 4; ++t) acc[t] = v8f{};

  for (int kk = 0; kk < K; kk += 32) {
    const v16bf a = ldbf(arow + kk + hi * 8, arow + kk + hi * 8 + 16);
    __builtin_prefetch(arow + kk + 64, 0, 1);        // two K-steps ahead (A stream)
    const int kb = kk + hi * 8;
#pragma unroll
    for (int t = 0; t < 4; ++t) {
      const __bf16* wp = wrow0 + (size_t)t * 16 * K + kb;
      v16bf b = ldbf(wp, wp + 16);
      __builtin_prefetch(wp + 32, 0, 1);             // next K-step (B stream)
      acc[t] = wmma_bf16(a, b, acc[t]);
    }
  }

#pragma unroll
  for (int t = 0; t < 4; ++t) {
    const int ncol = n0 + t * 16 + l16;
    const float bv = bias[ncol];
#pragma unroll
    for (int j = 0; j < 8; ++j) {
      const int m = mt * 16 + j + hi * 8;
      const float val = acc[t][j] + bv;
      if constexpr (MODE == 2) {
        ((float*)outp)[(size_t)m * N + ncol] = val;
      } else {
        const int b = m / Sc, s = m % Sc;
        const int h = ncol >> 6, dk = ncol & 63;
        if constexpr (MODE == 0)
          ((__bf16*)outp)[(((size_t)b * Hc + h) * Sc + s) * DKc + dk] = f2bf(val);
        else
          ((__bf16*)outp)[(((size_t)b * Hc + h) * DKc + dk) * Sc + s] = f2bf(val);
      }
    }
  }
}

// ---------------------------------------------------------------------------
// Fused attention: per (b,h, 32-row q tile)
//   phase1: scores(32x2048) = Q Kt / 8, mask -> LDS (fp32, 256KB of 320KB WGP)
//   phase2: row softmax in LDS, write attn to d_out (mandatory 537MB output)
//   phase3: X = attn @ V via WMMA (A from LDS, B from transposed V)
// ---------------------------------------------------------------------------
__global__ __launch_bounds__(256) void mha_attn_kernel(
    const __bf16* __restrict__ Qb, const __bf16* __restrict__ Kb,
    const __bf16* __restrict__ Vt, const int* __restrict__ mask,
    float* __restrict__ attn_out, __bf16* __restrict__ Xc) {
  extern __shared__ float sc[];                 // [32][2048]
  const int bid = blockIdx.x;
  const int qt  = bid & 63;                     // S/32 = 64 q-tiles
  const int bh  = bid >> 6;                     // 0..31
  const int b   = bh >> 4, h = bh & 15;
  const int lane = threadIdx.x & 31;
  const int w    = threadIdx.x >> 5;            // 8 waves
  const int l16  = lane & 15;
  const int hi   = lane >> 4;

  const __bf16* Qbase = Qb + (size_t)bh * Sc * DKc;
  const __bf16* Kbase = Kb + (size_t)bh * Sc * DKc;

  // ---- phase 1: scores ----
  v16bf afr[2][2];
#pragma unroll
  for (int qs = 0; qs < 2; ++qs)
#pragma unroll
    for (int ds = 0; ds < 2; ++ds) {
      const __bf16* p = Qbase + (size_t)(qt * 32 + qs * 16 + l16) * DKc +
                        ds * 32 + hi * 8;
      afr[qs][ds] = ldbf(p, p + 16);
    }

  const int k0w = w * 256;                      // this wave's key range
  for (int ks = 0; ks < 16; ++ks) {
    const int kc = k0w + ks * 16 + l16;         // key row (B-operand column)
    const __bf16* kp = Kbase + (size_t)kc * DKc;
    v16bf b0 = ldbf(kp + hi * 8,      kp + hi * 8 + 16);
    v16bf b1 = ldbf(kp + 32 + hi * 8, kp + 32 + hi * 8 + 16);
    const int mok = mask[b * Sc + kc];
#pragma unroll
    for (int qs = 0; qs < 2; ++qs) {
      v8f acc = v8f{};
      acc = wmma_bf16(afr[qs][0], b0, acc);
      acc = wmma_bf16(afr[qs][1], b1, acc);
#pragma unroll
      for (int j = 0; j < 8; ++j) {
        const int r = qs * 16 + j + hi * 8;
        sc[r * Sc + kc] = mok ? acc[j] * 0.125f : -1.0e9f;
      }
    }
  }
  __syncthreads();

  // ---- phase 2: softmax (wave w owns rows w*4 .. w*4+3) ----
  for (int i = 0; i < 4; ++i) {
    const int r = w * 4 + i;
    float* rowp = sc + r * Sc;
    float mx = -3.4e38f;
    for (int k = lane; k < Sc; k += 32) mx = fmaxf(mx, rowp[k]);
#pragma unroll
    for (int off = 16; off > 0; off >>= 1) mx = fmaxf(mx, __shfl_xor(mx, off, 32));
    float sum = 0.f;
    for (int k = lane; k < Sc; k += 32) {
      float e = __expf(rowp[k] - mx);
      rowp[k] = e;
      sum += e;
    }
#pragma unroll
    for (int off = 16; off > 0; off >>= 1) sum += __shfl_xor(sum, off, 32);
    const float inv = 1.0f / sum;
    float* op = attn_out + ((size_t)bh * Sc + qt * 32 + r) * Sc;
    for (int k = lane; k < Sc; k += 32) {
      float p = rowp[k] * inv;
      rowp[k] = p;
      op[k]   = p;
    }
  }
  __syncthreads();

  // ---- phase 3: attn @ V  (wave -> one 16x16 tile of the 32x64 output) ----
  const int qs   = w >> 2;                      // 0..1
  const int dsub = w & 3;                       // 0..3
  const __bf16* Vrow = Vt + (size_t)bh * DKc * Sc +
                       (size_t)(dsub * 16 + l16) * Sc;
  const float* sp = sc + (size_t)(qs * 16 + l16) * Sc;
  v8f acc = v8f{};
  for (int kt = 0; kt < 64; ++kt) {
    const int k1 = kt * 32 + hi * 8;
    v16bf a = cvtbf(sp + k1, sp + k1 + 16);     // probs fp32 -> bf16
    const int kb2 = kt * 32 + hi * 8;
    v16bf bfr = ldbf(Vrow + kb2, Vrow + kb2 + 16);
    __builtin_prefetch(Vrow + kb2 + 32, 0, 1);  // next K-step (V stream)
    acc = wmma_bf16(a, bfr, acc);
  }
#pragma unroll
  for (int j = 0; j < 8; ++j) {
    const int m    = qs * 16 + j + hi * 8;
    const int srow = qt * 32 + m;
    const int dk   = dsub * 16 + l16;
    Xc[((size_t)b * Sc + srow) * Dc + h * DKc + dk] = f2bf(acc[j]);
  }
}

// ---------------------------------------------------------------------------
extern "C" void kernel_launch(void* const* d_in, const int* in_sizes, int n_in,
                              void* d_out, int out_size, void* d_ws, size_t ws_size,
                              hipStream_t stream) {
  const float* query = (const float*)d_in[0];
  const float* key   = (const float*)d_in[1];
  const float* value = (const float*)d_in[2];
  const int*   mask  = (const int*)d_in[3];
  const float* Wq = (const float*)d_in[4],  *bq = (const float*)d_in[5];
  const float* Wk = (const float*)d_in[6],  *bk = (const float*)d_in[7];
  const float* Wv = (const float*)d_in[8],  *bv = (const float*)d_in[9];
  const float* Wo = (const float*)d_in[10], *bo = (const float*)d_in[11];

  float* out_y    = (float*)d_out;                         // (B,S,D)
  float* out_attn = (float*)d_out + (size_t)Bc * Sc * Dc;  // (B,H,S,S)

  // workspace layout (bf16):
  //   Qbf | Kbf | Vt | Xc           4 x (B*S*D)  = 32 MiB
  //   Qx  | Kx  | Vx (bf16 inputs)  3 x (B*S*D)  = 24 MiB
  //   Wqb | Wkb | Wvb | Wob         4 x (D*D)    =  8 MiB
  const size_t eBSD = (size_t)Bc * Sc * Dc;   // 4 Mi elements
  const size_t eDD  = (size_t)Dc * Dc;        // 1 Mi elements
  __bf16* Qbf = (__bf16*)d_ws;
  __bf16* Kbf = Qbf + eBSD;
  __bf16* Vt  = Kbf + eBSD;
  __bf16* Xc  = Vt  + eBSD;
  __bf16* Qx  = Xc  + eBSD;
  __bf16* Kx  = Qx  + eBSD;
  __bf16* Vx  = Kx  + eBSD;
  __bf16* Wqb = Vx  + eBSD;
  __bf16* Wkb = Wqb + eDD;
  __bf16* Wvb = Wkb + eDD;
  __bf16* Wob = Wvb + eDD;

  const dim3 thr(256);
  const dim3 cvtBSD((unsigned)(eBSD / 4 / 256));   // 4096 blocks
  const dim3 cvtDD((unsigned)(eDD / 4 / 256));     // 1024 blocks
  cvt_bf16_kernel<<<cvtBSD, thr, 0, stream>>>(query, Qx);
  cvt_bf16_kernel<<<cvtBSD, thr, 0, stream>>>(key,   Kx);
  cvt_bf16_kernel<<<cvtBSD, thr, 0, stream>>>(value, Vx);
  cvt_bf16_kernel<<<cvtDD,  thr, 0, stream>>>(Wq, Wqb);
  cvt_bf16_kernel<<<cvtDD,  thr, 0, stream>>>(Wk, Wkb);
  cvt_bf16_kernel<<<cvtDD,  thr, 0, stream>>>(Wv, Wvb);
  cvt_bf16_kernel<<<cvtDD,  thr, 0, stream>>>(Wo, Wob);

  const dim3 gblk(512);
  mha_gemm_kernel<0><<<gblk, thr, 0, stream>>>(Qx, Wqb, bq, Qbf);
  mha_gemm_kernel<0><<<gblk, thr, 0, stream>>>(Kx, Wkb, bk, Kbf);
  mha_gemm_kernel<1><<<gblk, thr, 0, stream>>>(Vx, Wvb, bv, Vt);

  const size_t lds_bytes = 32 * (size_t)Sc * sizeof(float);  // 256 KiB
  mha_attn_kernel<<<dim3(Bc * Hc * (Sc / 32)), thr, lds_bytes, stream>>>(
      Qbf, Kbf, Vt, mask, out_attn, Xc);

  mha_gemm_kernel<2><<<gblk, thr, 0, stream>>>(Xc, Wob, bo, out_y);
}
